// SceneGraph_36790689857543
// MI455X (gfx1250) — matrix operand order repaired and checked by reading, not compiled, not run on hardware
//
#include <hip/hip_runtime.h>
#include <hip/hip_bf16.h>

typedef __attribute__((ext_vector_type(16))) _Float16 v16h;
typedef __attribute__((ext_vector_type(8)))  float    v8f;

union F16x16 { v16h v; uint4 u[2]; };

__device__ __forceinline__ unsigned int pkh(float a, float b) {
    union { _Float16 h[2]; unsigned int u; } x;
    x.h[0] = (_Float16)a; x.h[1] = (_Float16)b;
    return x.u;
}
__device__ __forceinline__ unsigned short h16bits(float a) {
    union { _Float16 h; unsigned short u; } x;
    x.h = (_Float16)a;
    return x.u;
}

#define NN_   1536
#define DD_   768
#define NNSQ  (1536 * 1536)

// ---------------- convert E (1536x768 f32 -> f16, row major) ----------------
__global__ __launch_bounds__(256) void conv_e(const float* __restrict__ E,
                                              _Float16* __restrict__ Ef) {
    int id = blockIdx.x * 256 + threadIdx.x;          // 1536*768 = 4608*256
    Ef[id] = (_Float16)E[id];
}

// ---- convert weights: Wt[128][768] (n-major B for GEMM), w1sT[64][8], w2T[16][64] ----
__global__ __launch_bounds__(256) void conv_w(const float* __restrict__ W1,
                                              const float* __restrict__ W2,
                                              _Float16* __restrict__ Wt,
                                              _Float16* __restrict__ w1sT,
                                              _Float16* __restrict__ w2T) {
    int id = blockIdx.x * 256 + threadIdx.x;          // 99840 = 390*256
    if (id < 128 * 768) {
        int n = id / 768, k = id % 768;
        float v = (n < 64) ? W1[k * 64 + n] : W1[(768 + k) * 64 + (n - 64)];
        Wt[id] = (_Float16)v;
    } else if (id < 128 * 768 + 512) {
        int q = id - 128 * 768;                       // w1sT[n][s] = W1s[s][n]
        int n = q / 8, s = q % 8;
        w1sT[q] = (_Float16)W1[(1536 + s) * 64 + n];
    } else if (id < 128 * 768 + 512 + 1024) {
        int q = id - (128 * 768 + 512);               // w2T[n][k], n<4 valid
        int n = q / 64, k = q % 64;
        w2T[q] = (n < 4) ? (_Float16)W2[k * 4 + n] : (_Float16)0.0f;
    }
}

// ---------------- hab = E @ [W1a | W1b] : ha f32 [1536][64], hbT f16 [64][1536] ----------------
__global__ __launch_bounds__(32) void gemm_hab(const _Float16* __restrict__ Ef,
                                               const _Float16* __restrict__ Wt,
                                               float* __restrict__ ha,
                                               _Float16* __restrict__ hbT) {
    const int lane = threadIdx.x;
    const int h = lane >> 4, l = lane & 15;
    const int i0 = blockIdx.x * 16;
    const int n0 = blockIdx.y * 16;

    const uint4* arow = (const uint4*)(Ef + (i0 + l) * DD_);
    const uint4* brow = (const uint4*)(Wt + (n0 + l) * DD_);

    v8f acc = {};
    #pragma unroll 4
    for (int k = 0; k < DD_; k += 32) {
        F16x16 A, B;
        A.u[0] = arow[(k + h * 8) >> 3];        // K layout: lanes0-15 K0-7, lanes16-31 K8-15
        A.u[1] = arow[(k + 16 + h * 8) >> 3];   // then K16-23 / K24-31
        B.u[0] = brow[(k + h * 8) >> 3];        // B: lane = column, same K layout as A
        B.u[1] = brow[(k + 16 + h * 8) >> 3];
        acc = __builtin_amdgcn_wmma_f32_16x16x32_f16(false, A.v, false, B.v,
                                                     (short)0, acc, false, false);
    }

    if (n0 < 64) {                               // ha half: f32 row-major
        const int col = n0 + l;
        #pragma unroll
        for (int r = 0; r < 8; ++r)
            ha[(i0 + r + 8 * h) * 64 + col] = acc[r];
    } else {                                     // hb half: transposed f16 [n][j]
        const int n = n0 - 64 + l;
        uint4 p;
        p.x = pkh(acc[0], acc[1]); p.y = pkh(acc[2], acc[3]);
        p.z = pkh(acc[4], acc[5]); p.w = pkh(acc[6], acc[7]);
        *(uint4*)(hbT + n * NN_ + i0 + 8 * h) = p;
    }
}

// ---------------- pair kernel: 1 wave = 1 row i, 192-wide j strip ----------------
__global__ __launch_bounds__(32) void pair_kernel(const float* __restrict__ pos,
                                                  const float* __restrict__ prev,
                                                  const float* __restrict__ ha,
                                                  const _Float16* __restrict__ hbT,
                                                  const _Float16* __restrict__ w1sT,
                                                  const _Float16* __restrict__ w2T,
                                                  const float* __restrict__ b1,
                                                  const float* __restrict__ b2,
                                                  float* __restrict__ out) {
    __shared__ __align__(16) unsigned short Asp[32 * 32];   // A: [sp(8)|onehot(16)|0(8)] per row
    __shared__ __align__(16) unsigned short Hf[32 * 64];    // relu(hidden) f16, [pair][h]
    __shared__ __align__(16) float          Lg[32 * 4];     // logits, [pair][c]

    const int lane = threadIdx.x;
    const int h = lane >> 4, l = lane & 15;
    const int i = blockIdx.x;

    // --- per-i preloads ---
    const float pix = pos[i * 3 + 0], piy = pos[i * 3 + 1], piz = pos[i * 3 + 2];
    const float vix = pix - prev[i * 3 + 0];
    const float viy = piy - prev[i * 3 + 1];
    const float viz = piz - prev[i * 3 + 2];

    float haB[4];                                // C init: ha[i][col] + b1[col]
    #pragma unroll
    for (int t = 0; t < 4; ++t)
        haB[t] = ha[i * 64 + t * 16 + l] + b1[t * 16 + l];

    uint4 w1sPart[4];                            // W1s columns: B rows K0-7 (h==0 lanes)
    #pragma unroll
    for (int t = 0; t < 4; ++t)
        w1sPart[t] = *(const uint4*)(w1sT + (t * 16 + l) * 8);

    F16x16 w2f[2];                               // W2pad B fragments (64x16, 2 K-chunks)
    #pragma unroll
    for (int kk = 0; kk < 2; ++kk) {
        w2f[kk].u[0] = *(const uint4*)(w2T + l * 64 + kk * 32 + h * 8);
        w2f[kk].u[1] = *(const uint4*)(w2T + l * 64 + kk * 32 + 16 + h * 8);
    }
    const float b20 = b2[0], b21 = b2[1], b22 = b2[2], b23 = b2[3];

    // --- one-time A-matrix skeleton in LDS: zeros + one-hot identity block ---
    #pragma unroll
    for (int q = 0; q < 32; ++q) Asp[lane * 32 + q] = 0;
    Asp[lane * 32 + 8 + (lane & 15)] = 0x3C00;   // 1.0h at K = 8 + (row%16)
    __syncthreads();

    const int jstrip = blockIdx.y * 192;
    for (int it = 0; it < 6; ++it) {
        const int j0 = jstrip + it * 32;
        const int j = j0 + lane;

        // --- spatial features for this lane's pair (i, j) ---
        const float pjx = pos[j * 3 + 0], pjy = pos[j * 3 + 1], pjz = pos[j * 3 + 2];
        const float vjx = pjx - prev[j * 3 + 0];
        const float vjy = pjy - prev[j * 3 + 1];
        const float vjz = pjz - prev[j * 3 + 2];
        const float rx = pjx - pix, ry = pjy - piy, rz = pjz - piz;
        const float rvx = vjx - vix, rvy = vjy - viy, rvz = vjz - viz;
        const float dist = sqrtf(rx * rx + ry * ry + rz * rz);
        const float closing = -(rx * rvx + ry * rvy + rz * rvz) / fmaxf(dist, 1e-6f);
        const float vdiff = ry;

        uint4 spv;
        spv.x = pkh(rx, ry);  spv.y = pkh(rz, dist);
        spv.z = pkh(rvx, rvy); spv.w = pkh(rvz, vdiff);
        *(uint4*)(Asp + lane * 32) = spv;        // K0-7 of row `lane`
        __syncthreads();

        // --- hidden = relu(ha[i] + b1 + sp@W1s + hb[j]) via K-extended WMMA ---
        #pragma unroll
        for (int f = 0; f < 2; ++f) {
            F16x16 A;
            const int row = f * 16 + l;
            A.u[0] = *(const uint4*)(Asp + row * 32 + h * 8);
            A.u[1] = *(const uint4*)(Asp + row * 32 + 16 + h * 8);
            const int jb = j0 + f * 16;
            #pragma unroll
            for (int t = 0; t < 4; ++t) {
                const int n = t * 16 + l;
                const uint4 hbv = *(const uint4*)(hbT + n * NN_ + jb + 8 * (h ^ 1));
                F16x16 B;
                const uint4 z4 = make_uint4(0u, 0u, 0u, 0u);
                B.u[0] = h ? hbv : w1sPart[t];   // h=0: K0-7=W1s, K16-23=hb[8..15]
                B.u[1] = h ? z4 : hbv;           // h=1: K8-15=hb[0..7], K24-31=0
                v8f c;
                #pragma unroll
                for (int r = 0; r < 8; ++r) c[r] = haB[t];
                c = __builtin_amdgcn_wmma_f32_16x16x32_f16(false, A.v, false, B.v,
                                                           (short)0, c, false, false);
                #pragma unroll
                for (int r = 0; r < 8; ++r)
                    Hf[(f * 16 + r + 8 * h) * 64 + n] = h16bits(fmaxf(c[r], 0.0f));
            }
        }
        __syncthreads();

        // --- logits = hidden @ W2pad (2 WMMAs per 16-pair fragment) ---
        #pragma unroll
        for (int f = 0; f < 2; ++f) {
            v8f d = {};
            const int row = f * 16 + l;
            #pragma unroll
            for (int kk = 0; kk < 2; ++kk) {
                F16x16 A2;
                A2.u[0] = *(const uint4*)(Hf + row * 64 + kk * 32 + h * 8);
                A2.u[1] = *(const uint4*)(Hf + row * 64 + kk * 32 + 16 + h * 8);
                d = __builtin_amdgcn_wmma_f32_16x16x32_f16(false, A2.v, false, w2f[kk].v,
                                                           (short)0, d, false, false);
            }
            if (l < 4) {
                #pragma unroll
                for (int r = 0; r < 8; ++r)
                    Lg[(f * 16 + r + 8 * h) * 4 + l] = d[r];
            }
        }
        __syncthreads();

        // --- per-lane softmax + classification rules for pair (i, j0+lane) ---
        const float4 lgv = *(const float4*)(Lg + lane * 4);
        const float l0 = lgv.x + b20, l1 = lgv.y + b21, l2 = lgv.z + b22, l3 = lgv.w + b23;
        int   best = 0; float bm = l0;
        if (l1 > bm) { bm = l1; best = 1; }
        if (l2 > bm) { bm = l2; best = 2; }
        if (l3 > bm) { bm = l3; best = 3; }
        const float s = __expf(l0 - bm) + __expf(l1 - bm) + __expf(l2 - bm) + __expf(l3 - bm);
        const float conf = 1.0f / s;             // max prob = exp(0)/sum

        const bool near  = dist < 0.25f;
        const bool appr  = !near && (closing > 0.05f);
        const bool flee  = !near && !appr && (closing < -0.05f);
        const bool above = !near && !appr && !flee && (fabsf(vdiff) > 0.3f) && (dist < 0.5f);
        const int rel_type = near ? 0 : appr ? 1 : flee ? 2 : above ? 3 : best;
        const float conf_out = near  ? fmaxf(conf, 0.8f)
                             : appr  ? fmaxf(conf, 0.6f)
                             : flee  ? fmaxf(conf, 0.6f)
                             : above ? fmaxf(conf, 0.5f)
                             : conf;
        const bool valid = (j > i) && (conf_out > 0.3f);

        const int idx = i * NN_ + j;
        out[idx]            = (float)rel_type;
        out[NNSQ + idx]     = conf_out;
        out[2 * NNSQ + idx] = valid ? 1.0f : 0.0f;
        __syncthreads();
    }
}

extern "C" void kernel_launch(void* const* d_in, const int* in_sizes, int n_in,
                              void* d_out, int out_size, void* d_ws, size_t ws_size,
                              hipStream_t stream) {
    (void)in_sizes; (void)n_in; (void)out_size; (void)ws_size;
    const float* E    = (const float*)d_in[0];
    const float* pos  = (const float*)d_in[1];
    const float* prev = (const float*)d_in[2];
    const float* W1   = (const float*)d_in[3];
    const float* b1   = (const float*)d_in[4];
    const float* W2   = (const float*)d_in[5];
    const float* b2   = (const float*)d_in[6];

    char* ws = (char*)d_ws;
    _Float16* Ef   = (_Float16*)(ws + 0);          // 1536*768*2  = 2359296
    _Float16* Wt   = (_Float16*)(ws + 2359296);    // 128*768*2   = 196608
    float*    ha   = (float*)   (ws + 2555904);    // 1536*64*4   = 393216
    _Float16* hbT  = (_Float16*)(ws + 2949120);    // 64*1536*2   = 196608
    _Float16* w1sT = (_Float16*)(ws + 3145728);    // 64*8*2      = 1024
    _Float16* w2T  = (_Float16*)(ws + 3146752);    // 16*64*2     = 2048

    conv_e<<<4608, 256, 0, stream>>>(E, Ef);
    conv_w<<<390, 256, 0, stream>>>(W1, W2, Wt, w1sT, w2T);
    gemm_hab<<<dim3(96, 8), 32, 0, stream>>>(Ef, Wt, ha, hbT);
    pair_kernel<<<dim3(1536, 8), 32, 0, stream>>>(pos, prev, ha, hbT, w1sT, w2T,
                                                  b1, b2, (float*)d_out);
}
